// Net_10075993276849
// MI455X (gfx1250) — matrix operand order repaired and checked by reading, not compiled, not run on hardware
//
#include <hip/hip_runtime.h>

// Problem constants (match reference)
#define F_DIM 128
#define D_DIM 64
#define B_GRAPHS 128

typedef __attribute__((ext_vector_type(16))) __bf16 v16bf;
typedef __attribute__((ext_vector_type(8)))  float  v8f;

// ---------------- WMMA helpers (CDNA5 16x16x32 bf16, f32 accum) ----------------

__device__ __forceinline__ v8f wmma_bf16(v16bf a, v16bf b, v8f c) {
  // 8 args: (neg_a, A, neg_b, B, c_mod, C, reuse_a, reuse_b)
  return __builtin_amdgcn_wmma_f32_16x16x32_bf16(false, a, false, b, (short)0, c,
                                                 false, false);
}

// A-matrix 16x32 bf16 fragment (row = lane&15).
// Lanes 0-15 hold K = kbase+{0..7, 16..23}; lanes 16-31 hold K = kbase+{8..15, 24..31}.
__device__ __forceinline__ v16bf load_a_frag(const float* __restrict__ rowptr,
                                             int kbase, int hi) {
  const float* p0 = rowptr + kbase + hi * 8;
  const float* p1 = rowptr + kbase + 16 + hi * 8;
  float4 q0 = *(const float4*)(p0);
  float4 q1 = *(const float4*)(p0 + 4);
  float4 q2 = *(const float4*)(p1);
  float4 q3 = *(const float4*)(p1 + 4);
  v16bf a;
  a[0]  = (__bf16)q0.x; a[1]  = (__bf16)q0.y; a[2]  = (__bf16)q0.z; a[3]  = (__bf16)q0.w;
  a[4]  = (__bf16)q1.x; a[5]  = (__bf16)q1.y; a[6]  = (__bf16)q1.z; a[7]  = (__bf16)q1.w;
  a[8]  = (__bf16)q2.x; a[9]  = (__bf16)q2.y; a[10] = (__bf16)q2.z; a[11] = (__bf16)q2.w;
  a[12] = (__bf16)q3.x; a[13] = (__bf16)q3.y; a[14] = (__bf16)q3.z; a[15] = (__bf16)q3.w;
  return a;
}

// B-matrix 32x16 bf16 fragment from row-major W[K][ld].
// col = lane&15 (+n0); lanes 0-15 hold K = kbase+{0..7, 16..23}; lanes 16-31 hold K = kbase+{8..15, 24..31}.
__device__ __forceinline__ v16bf load_b_frag(const float* __restrict__ W, int ld,
                                             int kbase, int n0, int lane) {
  int c = n0 + (lane & 15);
  int k = kbase + (lane >> 4) * 8;
  const float* p = W + (long long)k * ld + c;
  v16bf b;
#pragma unroll
  for (int j = 0; j < 16; ++j) b[j] = (__bf16)p[((j & 7) + (j >> 3) * 16) * ld];
  return b;
}

// ---------------- Kernels ----------------

// deg[row[e]] += 1
__global__ void deg_count_kernel(const long long* __restrict__ rows,
                                 float* __restrict__ deg, long long E) {
  long long t = (long long)blockIdx.x * blockDim.x + threadIdx.x;
  if (t < E) atomicAdd(&deg[rows[t]], 1.0f);
}

// deg -> d^{-1/2} in place
__global__ void dinv_kernel(float* __restrict__ deg, int n) {
  int i = blockIdx.x * blockDim.x + threadIdx.x;
  if (i < n) {
    float d = deg[i];
    deg[i] = d > 0.f ? rsqrtf(d) : 0.f;
  }
}

// h = relu(x @ W_lin + b), x:[n,128], W:[128,64]. One wave -> 16x64 tile, 16 WMMAs.
__global__ void __launch_bounds__(256) gemm_lin_relu_kernel(
    const float* __restrict__ x, const float* __restrict__ Wl,
    const float* __restrict__ bias, float* __restrict__ h, int n_tiles) {
  const int lane = threadIdx.x & 31;
  const int wave = threadIdx.x >> 5;
  const int wpb  = blockDim.x >> 5;
  const int hi = lane >> 4;
  const int lc = lane & 15;

  // Hoist all 16 B fragments (weights are tiny; one load per wave)
  v16bf bf[4][4];
#pragma unroll
  for (int kt = 0; kt < 4; ++kt)
#pragma unroll
    for (int nt = 0; nt < 4; ++nt)
      bf[kt][nt] = load_b_frag(Wl, D_DIM, kt * 32, nt * 16, lane);

  for (int tile = blockIdx.x * wpb + wave; tile < n_tiles;
       tile += (int)gridDim.x * wpb) {
    const float* xrow = x + (long long)(tile * 16 + lc) * F_DIM;
    v8f acc[4];
#pragma unroll
    for (int nt = 0; nt < 4; ++nt) acc[nt] = (v8f){0,0,0,0,0,0,0,0};
#pragma unroll
    for (int kt = 0; kt < 4; ++kt) {
      v16bf a = load_a_frag(xrow, kt * 32, hi);
#pragma unroll
      for (int nt = 0; nt < 4; ++nt) acc[nt] = wmma_bf16(a, bf[kt][nt], acc[nt]);
    }
#pragma unroll
    for (int nt = 0; nt < 4; ++nt) {
      int col = nt * 16 + lc;
      float bv = bias[col];
#pragma unroll
      for (int v = 0; v < 8; ++v) {
        int r = tile * 16 + v + hi * 8;
        float o = acc[nt][v] + bv;
        h[(long long)r * D_DIM + col] = o > 0.f ? o : 0.f;
      }
    }
  }
}

// tx[col] += (-dinv[row]*dinv[col]) * h[row]; one thread = 4 features of 1 edge
__global__ void scatter_edges_kernel(const long long* __restrict__ ei,
                                     const float* __restrict__ dinv,
                                     const float* __restrict__ h,
                                     float* __restrict__ tx, long long E) {
  long long t = (long long)blockIdx.x * blockDim.x + threadIdx.x;
  if (t >= E * 16) return;
  long long e = t >> 4;
  int d0 = (int)(t & 15) * 4;
  long long r = ei[e];
  long long c = ei[E + e];
  float w = -dinv[r] * dinv[c];
  float4 hv = *(const float4*)(h + r * D_DIM + d0);
  float* dst = tx + c * D_DIM + d0;
  atomicAdd(dst + 0, w * hv.x);
  atomicAdd(dst + 1, w * hv.y);
  atomicAdd(dst + 2, w * hv.z);
  atomicAdd(dst + 3, w * hv.w);
}

// out = relu(h@W0 + tx@W1 + b), fused global_add_pool into g[B,64] via atomics.
__global__ void __launch_bounds__(256) gemm_cheb_pool_kernel(
    const float* __restrict__ h, const float* __restrict__ tx,
    const float* __restrict__ W0, const float* __restrict__ W1,
    const float* __restrict__ bias, const long long* __restrict__ batch,
    float* __restrict__ g, int n_tiles) {
  const int lane = threadIdx.x & 31;
  const int wave = threadIdx.x >> 5;
  const int wpb  = blockDim.x >> 5;
  const int hi = lane >> 4;
  const int lc = lane & 15;

  v16bf b0[2][4], b1[2][4];
#pragma unroll
  for (int kt = 0; kt < 2; ++kt)
#pragma unroll
    for (int nt = 0; nt < 4; ++nt) {
      b0[kt][nt] = load_b_frag(W0, D_DIM, kt * 32, nt * 16, lane);
      b1[kt][nt] = load_b_frag(W1, D_DIM, kt * 32, nt * 16, lane);
    }

  for (int tile = blockIdx.x * wpb + wave; tile < n_tiles;
       tile += (int)gridDim.x * wpb) {
    const long long rowbase = (long long)(tile * 16 + lc);
    v8f acc[4];
#pragma unroll
    for (int nt = 0; nt < 4; ++nt) acc[nt] = (v8f){0,0,0,0,0,0,0,0};
#pragma unroll
    for (int kt = 0; kt < 2; ++kt) {
      v16bf a = load_a_frag(h + rowbase * D_DIM, kt * 32, hi);
#pragma unroll
      for (int nt = 0; nt < 4; ++nt) acc[nt] = wmma_bf16(a, b0[kt][nt], acc[nt]);
    }
#pragma unroll
    for (int kt = 0; kt < 2; ++kt) {
      v16bf a = load_a_frag(tx + rowbase * D_DIM, kt * 32, hi);
#pragma unroll
      for (int nt = 0; nt < 4; ++nt) acc[nt] = wmma_bf16(a, b1[kt][nt], acc[nt]);
    }
    float bv[4];
#pragma unroll
    for (int nt = 0; nt < 4; ++nt) bv[nt] = bias[nt * 16 + lc];
#pragma unroll
    for (int v = 0; v < 8; ++v) {
      int r = tile * 16 + v + hi * 8;
      long long bb = batch[r];
      float* grow = g + bb * D_DIM;
#pragma unroll
      for (int nt = 0; nt < 4; ++nt) {
        float o = acc[nt][v] + bv[nt];
        o = o > 0.f ? o : 0.f;
        atomicAdd(grow + nt * 16 + lc, o);
      }
    }
  }
}

// pred[b] = [g1[b], g2[b]] . W_fc + b_fc
__global__ void fc_kernel(const float* __restrict__ g1, const float* __restrict__ g2,
                          const float* __restrict__ Wfc, const float* __restrict__ bfc,
                          float* __restrict__ out) {
  int b = threadIdx.x;
  if (b >= B_GRAPHS) return;
  float s = bfc[0];
#pragma unroll 4
  for (int j = 0; j < D_DIM; ++j) s += g1[b * D_DIM + j] * Wfc[j];
#pragma unroll 4
  for (int j = 0; j < D_DIM; ++j) s += g2[b * D_DIM + j] * Wfc[D_DIM + j];
  out[b] = s;
}

// ---------------- Launch ----------------

extern "C" void kernel_launch(void* const* d_in, const int* in_sizes, int n_in,
                              void* d_out, int out_size, void* d_ws, size_t ws_size,
                              hipStream_t stream) {
  const float*     x1   = (const float*)d_in[0];
  const float*     x2   = (const float*)d_in[1];
  const long long* ei1  = (const long long*)d_in[2];
  const long long* ei2  = (const long long*)d_in[3];
  const long long* bt1  = (const long long*)d_in[4];
  const long long* bt2  = (const long long*)d_in[5];
  const float*     Wl1  = (const float*)d_in[6];
  const float*     bl1  = (const float*)d_in[7];
  const float*     W0_1 = (const float*)d_in[8];
  const float*     W1_1 = (const float*)d_in[9];
  const float*     bc1  = (const float*)d_in[10];
  const float*     Wl2  = (const float*)d_in[11];
  const float*     bl2  = (const float*)d_in[12];
  const float*     W0_2 = (const float*)d_in[13];
  const float*     W1_2 = (const float*)d_in[14];
  const float*     bc2  = (const float*)d_in[15];
  const float*     Wfc  = (const float*)d_in[16];
  const float*     bfc  = (const float*)d_in[17];

  const int       n       = in_sizes[4];          // nodes
  const long long E       = in_sizes[2] / 2;      // edges
  const int       n_tiles = n / 16;               // N=100000 -> 6250

  // Workspace layout (sequential reuse across encoders): ~52 MB
  float* ws  = (float*)d_ws;
  float* h   = ws;                                   // n*64
  float* tx  = ws + (size_t)n * D_DIM;               // n*64
  float* deg = ws + (size_t)2 * n * D_DIM;           // n
  float* g1  = deg + n;                              // B*64
  float* g2  = g1 + B_GRAPHS * D_DIM;                // B*64

  const int TPB = 256;
  const int gemm_blocks    = (n_tiles + 7) / 8;                       // 8 waves/block
  const int deg_blocks     = (int)((E + TPB - 1) / TPB);
  const int dinv_blocks    = (n + TPB - 1) / TPB;
  const int scatter_blocks = (int)((E * 16 + TPB - 1) / TPB);

  for (int enc = 0; enc < 2; ++enc) {
    const float*     x    = enc ? x2 : x1;
    const long long* ei   = enc ? ei2 : ei1;
    const long long* bt   = enc ? bt2 : bt1;
    const float*     Wl   = enc ? Wl2 : Wl1;
    const float*     bl   = enc ? bl2 : bl1;
    const float*     W0   = enc ? W0_2 : W0_1;
    const float*     W1   = enc ? W1_2 : W1_1;
    const float*     bc   = enc ? bc2 : bc1;
    float*           g    = enc ? g2 : g1;

    hipMemsetAsync(deg, 0, (size_t)n * sizeof(float), stream);
    hipMemsetAsync(tx, 0, (size_t)n * D_DIM * sizeof(float), stream);
    hipMemsetAsync(g, 0, (size_t)B_GRAPHS * D_DIM * sizeof(float), stream);

    deg_count_kernel<<<deg_blocks, TPB, 0, stream>>>(ei, deg, E);
    dinv_kernel<<<dinv_blocks, TPB, 0, stream>>>(deg, n);
    gemm_lin_relu_kernel<<<gemm_blocks, TPB, 0, stream>>>(x, Wl, bl, h, n_tiles);
    scatter_edges_kernel<<<scatter_blocks, TPB, 0, stream>>>(ei, deg, h, tx, E);
    gemm_cheb_pool_kernel<<<gemm_blocks, TPB, 0, stream>>>(h, tx, W0, W1, bc, bt, g,
                                                           n_tiles);
  }

  fc_kernel<<<1, B_GRAPHS, 0, stream>>>(g1, g2, Wfc, bfc, (float*)d_out);
}
